// ATMT5CE4Attention_29549374996520
// MI455X (gfx1250) — matrix operand order repaired and checked by reading, not compiled, not run on hardware
//
#include <hip/hip_runtime.h>
#include <math.h>
#include <stdint.h>

// ---------------------------------------------------------------------------
// Problem constants (T5 encoder self-attention)
// ---------------------------------------------------------------------------
#define BATCH      2
#define SEQ        2048
#define DMODEL     1024
#define NHEADS     16
#define HEADDIM    64
#define NBUCKETS   32
#define MAXDIST    128
#define SCALING    0.125f            // 64^-0.5
#define MROWS      (BATCH * SEQ)     // 4096

typedef float  v8f  __attribute__((ext_vector_type(8)));
typedef __bf16 v8bf __attribute__((ext_vector_type(8)));
typedef __bf16 v16bf __attribute__((ext_vector_type(16)));
typedef int    v4i  __attribute__((ext_vector_type(4)));

// Build a 16-element bf16 WMMA fragment from two contiguous 16B runs.
__device__ __forceinline__ v16bf frag16(const __bf16* p0, const __bf16* p1) {
    v8bf a = *(const v8bf*)p0;
    v8bf b = *(const v8bf*)p1;
    return __builtin_shufflevector(a, b, 0,1,2,3,4,5,6,7,8,9,10,11,12,13,14,15);
}

__device__ __forceinline__ void split_bf16(float x, __bf16& h, __bf16& l) {
    h = (__bf16)x;
    l = (__bf16)(x - (float)h);
}

#define WMMA_BF16(A, B, C) \
    __builtin_amdgcn_wmma_f32_16x16x32_bf16(false, (A), false, (B), (short)0, (C), false, false)

// ---------------------------------------------------------------------------
// CDNA5 async global->LDS staging (ASYNCcnt-tracked DMA, no VGPR round-trip)
// Builtin signature (from compiler diagnostic): (v4i AS1*, v4i AS3*, imm, imm)
// ---------------------------------------------------------------------------
__device__ __forceinline__ void async_load_b128(const __bf16* g, __bf16* l) {
#if __has_builtin(__builtin_amdgcn_global_load_async_to_lds_b128)
    __builtin_amdgcn_global_load_async_to_lds_b128(
        (__attribute__((address_space(1))) v4i*)(uintptr_t)g,
        (__attribute__((address_space(3))) v4i*)l,
        0, 0);
#else
    unsigned int loff =
        (unsigned int)(uintptr_t)(__attribute__((address_space(3))) void*)l;
    asm volatile("global_load_async_to_lds_b128 %0, %1, off"
                 :: "v"(loff), "v"((unsigned long long)(uintptr_t)g)
                 : "memory");
#endif
}

template <int N>
__device__ __forceinline__ void async_wait() {
#if __has_builtin(__builtin_amdgcn_s_wait_asynccnt)
    __builtin_amdgcn_s_wait_asynccnt(N);
#else
    asm volatile("s_wait_asynccnt %0" :: "i"(N) : "memory");
#endif
}

// ---------------------------------------------------------------------------
// Kernel: split fp32 array into bf16 hi/lo pair (x = hi + lo)
// ---------------------------------------------------------------------------
__global__ __launch_bounds__(256)
void split_kernel(const float* __restrict__ src,
                  __bf16* __restrict__ hi, __bf16* __restrict__ lo, int n)
{
    int i = blockIdx.x * 256 + threadIdx.x;
    if (i < n) {
        __bf16 h, l;
        split_bf16(src[i], h, l);
        hi[i] = h; lo[i] = l;
    }
}

// ---------------------------------------------------------------------------
// Kernel: per-head relative-position bias table.
// biasTab[h][dist + 2047], dist = k - q in [-2047, 2047], padded to 4096.
// ---------------------------------------------------------------------------
__global__ __launch_bounds__(256)
void t5_bias_table_kernel(const float* __restrict__ rel_emb,  // [32][16]
                          float* __restrict__ biasTab)        // [16][4096]
{
    int idx = blockIdx.x * blockDim.x + threadIdx.x;
    if (idx >= NHEADS * 4096) return;
    int h = idx >> 12;
    int d = idx & 4095;
    if (d >= 4095) { biasTab[idx] = 0.0f; return; }

    int rp = d - 2047;                         // memory_pos - context_pos
    const int nb = NBUCKETS / 2;               // 16 (bidirectional)
    int bucket = (rp > 0) ? nb : 0;
    int arp = rp < 0 ? -rp : rp;
    const int max_exact = nb / 2;              // 8
    if (arp < max_exact) {
        bucket += arp;
    } else {
        float t = logf((float)arp * (1.0f / (float)max_exact)) *
                  (1.0f / logf((float)MAXDIST / (float)max_exact)) *
                  (float)(nb - max_exact);
        int large = max_exact + (int)t;
        bucket += (large < nb - 1) ? large : (nb - 1);
    }
    biasTab[idx] = rel_emb[bucket * NHEADS + h];
}

// ---------------------------------------------------------------------------
// Kernel: GEMM  Y[M][N] = X[M][K] @ W[N][K]^T  via bf16x3 split products.
// 128 threads = 4 waves; block tile 64x64; wave tile 32x32 (2x2 WMMA tiles);
// K in chunks of 32; double-buffered async global->LDS staging.
// mode 0: f32 Y.  mode 1: bf16 hi/lo Y [M][N].  mode 2: bf16 hi/lo [b][h][d][s].
// ---------------------------------------------------------------------------
__global__ __launch_bounds__(128)
void gemm_bf16x3_kernel(const __bf16* __restrict__ Xh, const __bf16* __restrict__ Xl,
                        const __bf16* __restrict__ Wh, const __bf16* __restrict__ Wl,
                        float* __restrict__ Yf,
                        __bf16* __restrict__ Yh, __bf16* __restrict__ Yl,
                        int M, int N, int K, int mode)
{
    // rows padded to 40 elems = 80B (16B multiple, 20-bank stride: aligned,
    // conflict-free b128 fragment reads). Double buffered.
    __shared__ __attribute__((aligned(16))) __bf16 Ash[2][64][40];
    __shared__ __attribute__((aligned(16))) __bf16 Asl[2][64][40];
    __shared__ __attribute__((aligned(16))) __bf16 Bsh[2][64][40];
    __shared__ __attribute__((aligned(16))) __bf16 Bsl[2][64][40];

    const int tid  = threadIdx.x;
    const int lane = tid & 31;
    const int wave = tid >> 5;
    const int m0 = blockIdx.y * 64;
    const int n0 = blockIdx.x * 64;
    const int wm = (wave & 1) * 32;
    const int wn = (wave >> 1) * 32;
    const int lm   = lane & 15;
    const int half = lane >> 4;

    // stage one 64x32 K-chunk (A rows from X, B rows from W), 8 async ops/lane
    auto stage = [&](int buf, int k0) {
        #pragma unroll
        for (int f = tid; f < 256; f += 128) {
            int row = f >> 2;
            int c8  = (f & 3) * 8;
            size_t ga = (size_t)(m0 + row) * K + k0 + c8;
            size_t gb = (size_t)(n0 + row) * K + k0 + c8;
            async_load_b128(Xh + ga, &Ash[buf][row][c8]);
            async_load_b128(Xl + ga, &Asl[buf][row][c8]);
            async_load_b128(Wh + gb, &Bsh[buf][row][c8]);
            async_load_b128(Wl + gb, &Bsl[buf][row][c8]);
        }
    };

    v8f c[2][2];
    #pragma unroll
    for (int i = 0; i < 2; ++i)
        #pragma unroll
        for (int j = 0; j < 2; ++j)
            c[i][j] = (v8f){0.f,0.f,0.f,0.f,0.f,0.f,0.f,0.f};

    stage(0, 0);
    int cur = 0;
    for (int k0 = 0; k0 < K; k0 += 32, cur ^= 1) {
        if (k0 + 32 < K) {
            stage(cur ^ 1, k0 + 32);   // prefetch next chunk into other buffer
            async_wait<8>();           // in-order: <=8 outstanding => cur landed
        } else {
            async_wait<0>();
        }
        __syncthreads();

        // Fragments per ISA 16-bit layouts:
        //  A lane: runs at [8*half, +8) and [16+8*half, +8)
        //  B lane: runs at [8*half, +8) and [16+8*half, +8)   (B col n == W row n: no transpose)
        v16bf ah[2], al[2], bh[2], bl[2];
        #pragma unroll
        for (int ti = 0; ti < 2; ++ti) {
            const __bf16* rh = &Ash[cur][wm + ti * 16 + lm][0];
            const __bf16* rl = &Asl[cur][wm + ti * 16 + lm][0];
            ah[ti] = frag16(rh + 8 * half, rh + 16 + 8 * half);
            al[ti] = frag16(rl + 8 * half, rl + 16 + 8 * half);
        }
        #pragma unroll
        for (int tj = 0; tj < 2; ++tj) {
            const __bf16* rh = &Bsh[cur][wn + tj * 16 + lm][0];
            const __bf16* rl = &Bsl[cur][wn + tj * 16 + lm][0];
            bh[tj] = frag16(rh + 8 * half, rh + 16 + 8 * half);
            bl[tj] = frag16(rl + 8 * half, rl + 16 + 8 * half);
        }
        #pragma unroll
        for (int ti = 0; ti < 2; ++ti)
            #pragma unroll
            for (int tj = 0; tj < 2; ++tj) {
                c[ti][tj] = WMMA_BF16(ah[ti], bh[tj], c[ti][tj]);
                c[ti][tj] = WMMA_BF16(ah[ti], bl[tj], c[ti][tj]);
                c[ti][tj] = WMMA_BF16(al[ti], bh[tj], c[ti][tj]);
            }
        __syncthreads();   // all reads of 'cur' done before it is re-staged
    }

    // Epilogue: C layout -> VGPR r = row r (lanes 0-15) / row r+8 (lanes 16-31)
    #pragma unroll
    for (int ti = 0; ti < 2; ++ti)
        #pragma unroll
        for (int tj = 0; tj < 2; ++tj)
            #pragma unroll
            for (int r = 0; r < 8; ++r) {
                int row = m0 + wm + ti * 16 + r + half * 8;
                int col = n0 + wn + tj * 16 + lm;
                float val = c[ti][tj][r];
                if (mode == 0) {
                    Yf[(size_t)row * N + col] = val;
                } else {
                    __bf16 h, l;
                    split_bf16(val, h, l);
                    if (mode == 1) {
                        size_t off = (size_t)row * N + col;
                        Yh[off] = h; Yl[off] = l;
                    } else {  // per-head transpose: [b][h][d][s]
                        int bb = row >> 11, s = row & 2047;
                        int hh = col >> 6,  d = col & 63;
                        size_t off = (((size_t)(bb * NHEADS + hh) * HEADDIM + d) * SEQ) + s;
                        Yh[off] = h; Yl[off] = l;
                    }
                }
            }
}

// ---------------------------------------------------------------------------
// Kernel: flash attention, bf16x3 WMMA, fp32 online softmax.
// 4 waves/block, wave owns 16 q rows; 32-key tiles, double-buffered async
// K/V staging; per-block bias window cached in LDS (2112 reachable distances).
// ---------------------------------------------------------------------------
__global__ __launch_bounds__(128)
void t5_flash_attn_kernel(const __bf16* __restrict__ Qh, const __bf16* __restrict__ Ql,
                          const __bf16* __restrict__ Kh, const __bf16* __restrict__ Kl,
                          const __bf16* __restrict__ Vth, const __bf16* __restrict__ Vtl,
                          const float* __restrict__ biasTab,   // [16][4096]
                          __bf16* __restrict__ Ch, __bf16* __restrict__ Cl)
{
    __shared__ float biasSh[2112];
    __shared__ __attribute__((aligned(16))) __bf16 Ksh[2][32][72];
    __shared__ __attribute__((aligned(16))) __bf16 Ksl[2][32][72];
    __shared__ __attribute__((aligned(16))) __bf16 Vsh[2][64][40];
    __shared__ __attribute__((aligned(16))) __bf16 Vsl[2][64][40];
    __shared__ __attribute__((aligned(16))) __bf16 Psh[4][16][40];
    __shared__ __attribute__((aligned(16))) __bf16 Psl[4][16][40];

    const int tid  = threadIdx.x;
    const int lane = tid & 31;
    const int wave = tid >> 5;
    const int h    = blockIdx.y;
    const int b    = blockIdx.z;
    const int qb   = blockIdx.x * 64;
    const int q0   = qb + wave * 16;

    const int lm   = lane & 15;
    const int half = lane >> 4;

    // Bias window: dist = k - q for q in [qb, qb+64), k in [0, 2048)
    // -> global idx [1984 - qb, 1984 - qb + 2111]
    const int g0 = 1984 - qb;
    for (int i = tid; i < 2112; i += 128)
        biasSh[i] = biasTab[(size_t)h * 4096 + g0 + i];

    const __bf16* Kbh = Kh + (size_t)(b * SEQ) * DMODEL + (size_t)h * HEADDIM;
    const __bf16* Kbl = Kl + (size_t)(b * SEQ) * DMODEL + (size_t)h * HEADDIM;
    const __bf16* Vbh = Vth + (size_t)(b * NHEADS + h) * HEADDIM * SEQ;
    const __bf16* Vbl = Vtl + (size_t)(b * NHEADS + h) * HEADDIM * SEQ;

    // stage one 32-key tile of K (row-major) and V^T (dim-major): 8 async/lane
    auto stage = [&](int buf, int kt) {
        #pragma unroll
        for (int f = tid; f < 256; f += 128) {
            int krow = f >> 3, kc8 = (f & 7) * 8;
            size_t gk = (size_t)(kt + krow) * DMODEL + kc8;
            async_load_b128(Kbh + gk, &Ksh[buf][krow][kc8]);
            async_load_b128(Kbl + gk, &Ksl[buf][krow][kc8]);
            int vrow = f >> 2, vc8 = (f & 3) * 8;
            size_t gv = (size_t)vrow * SEQ + kt + vc8;
            async_load_b128(Vbh + gv, &Vsh[buf][vrow][vc8]);
            async_load_b128(Vbl + gv, &Vsl[buf][vrow][vc8]);
        }
    };

    // Q tile 16x64 in registers as WMMA-A fragments (2 d-chunks, hi+lo)
    v16bf qh[2], ql[2];
    {
        const __bf16* qrh = Qh + (size_t)(b * SEQ + q0 + lm) * DMODEL + (size_t)h * HEADDIM;
        const __bf16* qrl = Ql + (size_t)(b * SEQ + q0 + lm) * DMODEL + (size_t)h * HEADDIM;
        #pragma unroll
        for (int kc = 0; kc < 2; ++kc) {
            qh[kc] = frag16(qrh + kc * 32 + 8 * half, qrh + kc * 32 + 16 + 8 * half);
            ql[kc] = frag16(qrl + kc * 32 + 8 * half, qrl + kc * 32 + 16 + 8 * half);
        }
    }

    v8f o[4];
    #pragma unroll
    for (int j = 0; j < 4; ++j) o[j] = (v8f){0.f,0.f,0.f,0.f,0.f,0.f,0.f,0.f};
    float mstat[8], lstat[8];
    #pragma unroll
    for (int r = 0; r < 8; ++r) { mstat[r] = -INFINITY; lstat[r] = 0.0f; }

    stage(0, 0);
    int cur = 0;
    for (int kt = 0; kt < SEQ; kt += 32, cur ^= 1) {
        if (kt + 32 < SEQ) {
            stage(cur ^ 1, kt + 32);
            async_wait<8>();
        } else {
            async_wait<0>();
        }
        __syncthreads();

        // ---- S = Q.K^T : 2 score n-tiles x 2 d-chunks x 3 split terms ----
        v8f s0 = (v8f){0.f,0.f,0.f,0.f,0.f,0.f,0.f,0.f};
        v8f s1 = s0;
        #pragma unroll
        for (int kc = 0; kc < 2; ++kc) {
            const __bf16* r0h = &Ksh[cur][lm     ][kc * 32 + 8 * half];
            const __bf16* r0l = &Ksl[cur][lm     ][kc * 32 + 8 * half];
            const __bf16* r1h = &Ksh[cur][16 + lm][kc * 32 + 8 * half];
            const __bf16* r1l = &Ksl[cur][16 + lm][kc * 32 + 8 * half];
            v16bf b0h = frag16(r0h, r0h + 16), b0l = frag16(r0l, r0l + 16);
            v16bf b1h = frag16(r1h, r1h + 16), b1l = frag16(r1l, r1l + 16);
            s0 = WMMA_BF16(qh[kc], b0h, s0);
            s0 = WMMA_BF16(qh[kc], b0l, s0);
            s0 = WMMA_BF16(ql[kc], b0h, s0);
            s1 = WMMA_BF16(qh[kc], b1h, s1);
            s1 = WMMA_BF16(qh[kc], b1l, s1);
            s1 = WMMA_BF16(ql[kc], b1h, s1);
        }

        // ---- online softmax over 32 keys (row = r + half*8) ----
        #pragma unroll
        for (int r = 0; r < 8; ++r) {
            int loc = kt + lm + 63 - wave * 16 - r - half * 8;   // bias index
            float sc0 = s0[r] * SCALING + biasSh[loc];
            float sc1 = s1[r] * SCALING + biasSh[loc + 16];
            float mx = fmaxf(sc0, sc1);
            #pragma unroll
            for (int off = 8; off; off >>= 1)
                mx = fmaxf(mx, __shfl_xor(mx, off, 16));
            float mnew  = fmaxf(mstat[r], mx);
            float alpha = __expf(mstat[r] - mnew);
            float p0    = __expf(sc0 - mnew);
            float p1    = __expf(sc1 - mnew);
            float rs    = p0 + p1;
            #pragma unroll
            for (int off = 8; off; off >>= 1)
                rs += __shfl_xor(rs, off, 16);
            lstat[r] = lstat[r] * alpha + rs;
            mstat[r] = mnew;
            #pragma unroll
            for (int j = 0; j < 4; ++j) o[j][r] *= alpha;
            __bf16 p0h, p0l, p1h, p1l;
            split_bf16(p0, p0h, p0l);
            split_bf16(p1, p1h, p1l);
            Psh[wave][r + half * 8][lm]      = p0h;
            Psl[wave][r + half * 8][lm]      = p0l;
            Psh[wave][r + half * 8][16 + lm] = p1h;
            Psl[wave][r + half * 8][16 + lm] = p1l;
        }
        asm volatile("" ::: "memory");   // keep wave-local DS store->load order

        // ---- O += P.V : K=32 chunk x 3 terms x 4 output dim-tiles ----
        {
            const __bf16* prh = &Psh[wave][lm][0];
            const __bf16* prl = &Psl[wave][lm][0];
            v16bf pah = frag16(prh + 8 * half, prh + 16 + 8 * half);
            v16bf pal = frag16(prl + 8 * half, prl + 16 + 8 * half);
            #pragma unroll
            for (int jt = 0; jt < 4; ++jt) {
                const __bf16* vrh = &Vsh[cur][jt * 16 + lm][8 * half];
                const __bf16* vrl = &Vsl[cur][jt * 16 + lm][8 * half];
                v16bf vbh = frag16(vrh, vrh + 16);
                v16bf vbl = frag16(vrl, vrl + 16);
                o[jt] = WMMA_BF16(pah, vbh, o[jt]);
                o[jt] = WMMA_BF16(pah, vbl, o[jt]);
                o[jt] = WMMA_BF16(pal, vbh, o[jt]);
            }
        }
        __syncthreads();   // all reads of 'cur' done before it is re-staged
    }

    // ---- finalize: ctx = O / l as bf16 hi/lo for the O-projection GEMM ----
    size_t cbase = (size_t)(b * SEQ + q0) * DMODEL + (size_t)h * HEADDIM;
    #pragma unroll
    for (int r = 0; r < 8; ++r) {
        float inv = 1.0f / lstat[r];
        #pragma unroll
        for (int j = 0; j < 4; ++j) {
            float val = o[j][r] * inv;
            __bf16 hh, ll;
            split_bf16(val, hh, ll);
            size_t off = cbase + (size_t)(r + half * 8) * DMODEL + j * 16 + lm;
            Ch[off] = hh; Cl[off] = ll;
        }
    }
}

// ---------------------------------------------------------------------------
// Host-side orchestration
// ---------------------------------------------------------------------------
extern "C" void kernel_launch(void* const* d_in, const int* in_sizes, int n_in,
                              void* d_out, int out_size, void* d_ws, size_t ws_size,
                              hipStream_t stream) {
    const float* X    = (const float*)d_in[0];   // [2,2048,1024]
    const float* Wq   = (const float*)d_in[1];   // [1024,1024]
    const float* Wk   = (const float*)d_in[2];
    const float* Wv   = (const float*)d_in[3];
    const float* Wo   = (const float*)d_in[4];
    const float* rel  = (const float*)d_in[5];   // [32,16]

    const size_t MAT = (size_t)MROWS * DMODEL;   // 4096*1024 elements
    const size_t WMT = (size_t)DMODEL * DMODEL;  // 1024*1024 elements

    __bf16* p = (__bf16*)d_ws;
    __bf16 *Xh = p, *Xl = p + MAT;                      p += 2 * MAT;
    __bf16 *Wqh = p, *Wql = p + WMT;                    p += 2 * WMT;
    __bf16 *Wkh = p, *Wkl = p + WMT;                    p += 2 * WMT;
    __bf16 *Wvh = p, *Wvl = p + WMT;                    p += 2 * WMT;
    __bf16 *Woh = p, *Wol = p + WMT;                    p += 2 * WMT;
    __bf16 *Qh = p, *Ql = p + MAT;                      p += 2 * MAT;
    __bf16 *Kh = p, *Kl = p + MAT;                      p += 2 * MAT;
    __bf16 *Vth = p, *Vtl = p + MAT;                    p += 2 * MAT;
    __bf16 *Ch = p, *Cl = p + MAT;                      p += 2 * MAT;
    float* bias = (float*)p;                            // 16*4096 floats

    // bf16 hi/lo decompositions
    split_kernel<<<(int)((MAT + 255) / 256), 256, 0, stream>>>(X,  Xh,  Xl,  (int)MAT);
    split_kernel<<<(int)((WMT + 255) / 256), 256, 0, stream>>>(Wq, Wqh, Wql, (int)WMT);
    split_kernel<<<(int)((WMT + 255) / 256), 256, 0, stream>>>(Wk, Wkh, Wkl, (int)WMT);
    split_kernel<<<(int)((WMT + 255) / 256), 256, 0, stream>>>(Wv, Wvh, Wvl, (int)WMT);
    split_kernel<<<(int)((WMT + 255) / 256), 256, 0, stream>>>(Wo, Woh, Wol, (int)WMT);

    // bias table
    t5_bias_table_kernel<<<(NHEADS * 4096 + 255) / 256, 256, 0, stream>>>(rel, bias);

    // projections (Q,K row-major bf16 pairs; V transposed per head)
    dim3 ggrid(DMODEL / 64, MROWS / 64);         // (16, 64)
    gemm_bf16x3_kernel<<<ggrid, 128, 0, stream>>>(Xh, Xl, Wqh, Wql,
        nullptr, Qh, Ql, MROWS, DMODEL, DMODEL, 1);
    gemm_bf16x3_kernel<<<ggrid, 128, 0, stream>>>(Xh, Xl, Wkh, Wkl,
        nullptr, Kh, Kl, MROWS, DMODEL, DMODEL, 1);
    gemm_bf16x3_kernel<<<ggrid, 128, 0, stream>>>(Xh, Xl, Wvh, Wvl,
        nullptr, Vth, Vtl, MROWS, DMODEL, DMODEL, 2);

    // flash attention: grid (S/64, H, B), 4 waves per block
    dim3 agrid(SEQ / 64, NHEADS, BATCH);
    t5_flash_attn_kernel<<<agrid, 128, 0, stream>>>(Qh, Ql, Kh, Kl, Vth, Vtl,
                                                    bias, Ch, Cl);

    // output projection -> d_out (fp32)
    gemm_bf16x3_kernel<<<ggrid, 128, 0, stream>>>(Ch, Cl, Woh, Wol,
        (float*)d_out, nullptr, nullptr, MROWS, DMODEL, DMODEL, 0);
}
